// MPNN_ENN_Sum_22153441313212
// MI455X (gfx1250) — matrix operand order repaired and checked by reading, not compiled, not run on hardware
//
#include <hip/hip_runtime.h>
#include <hip/hip_bf16.h>

// ---------------------------------------------------------------------------
// MPNN with edge-network messages (ENN), GRU update, sum-pool.
// N=12800 nodes, E=50000 edges, H=64, 12 steps, B=128 graphs.
//
// Cost model (MI455X, 23.3 TB/s HBM):
//   - A tensor (E x 64 x 64) in bf16 = 410 MB, computed ONCE with WMMA.
//   - 12 steps x stream A = 4.9 GB  => ~0.21 ms, the dominant term.
//   - everything else is << 100 us.
// ---------------------------------------------------------------------------

typedef __attribute__((ext_vector_type(16))) __bf16 v16bf;
typedef __attribute__((ext_vector_type(8)))  float  v8f;

#define N_NODES 12800
#define N_EDGES 50000
#define NFEAT   16
#define EFEAT   8
#define HDIM    64
#define BGRAPH  128
#define NSTEPS  12

// --------------------------------------------------------------------------
// h0 = node_features @ W_in + b_in        [N,16]@[16,64] -> [N,64]
// --------------------------------------------------------------------------
__global__ void __launch_bounds__(256) node_in_kernel(
    const float* __restrict__ nf, const float* __restrict__ Win,
    const float* __restrict__ bin, float* __restrict__ h) {
  int idx = blockIdx.x * 256 + threadIdx.x;          // n*64 + d
  if (idx >= N_NODES * HDIM) return;
  int n = idx >> 6, d = idx & 63;
  float acc = bin[d];
  const float* nr = nf + n * NFEAT;
#pragma unroll
  for (int k = 0; k < NFEAT; ++k) acc = fmaf(nr[k], Win[k * HDIM + d], acc);
  h[idx] = acc;
}

// --------------------------------------------------------------------------
// Transpose+convert ee_W2 [64,4096] f32 -> W2T [4096,64] bf16
// (row-major per output column => contiguous K for WMMA A-fragments)
// --------------------------------------------------------------------------
__global__ void __launch_bounds__(256) w2t_kernel(
    const float* __restrict__ W2, __bf16* __restrict__ W2T) {
  int idx = blockIdx.x * 256 + threadIdx.x;          // n*64 + k, 262144 total
  int n = idx >> 6, k = idx & 63;
  W2T[idx] = (__bf16)W2[k * 4096 + n];
}

// --------------------------------------------------------------------------
// Edge encoder: eh = relu(ef @ W1 + b1); A = eh @ W2 + b2  -> bf16 [E,64,64]
// One block handles 16 edges. WMMA roles:
//   A-matrix  = W2T tile   (M = 16 output columns, K = 64)
//   B-matrix  = eh tile^T  (K = 64, N = 16 edges)
//   C 16x16 f32: lane -> edge (N), VGPR -> column (M)  => packed b128 stores
// Emits v_wmma_f32_16x16x32_bf16 (2 per column tile, 512 per block).
// --------------------------------------------------------------------------
__global__ void __launch_bounds__(256) edge_enc_kernel(
    const float* __restrict__ ef, const float* __restrict__ W1,
    const float* __restrict__ b1, const __bf16* __restrict__ W2T,
    const float* __restrict__ b2, __bf16* __restrict__ A) {
  __shared__ __bf16 ehs[16 * 64];                    // eh tile, row-major [edge][k]
  const int e0 = blockIdx.x * 16;
  const int t  = threadIdx.x;

  // ---- phase 1: eh tile (1024 values, 4 per thread), relu, store bf16 ----
#pragma unroll
  for (int i = 0; i < 4; ++i) {
    int idx = t + i * 256;                           // local_edge*64 + d
    int le = idx >> 6, d = idx & 63;
    float acc = b1[d];
    const float* er = ef + (size_t)(e0 + le) * EFEAT;
#pragma unroll
    for (int k = 0; k < EFEAT; ++k) acc = fmaf(er[k], W1[k * HDIM + d], acc);
    ehs[idx] = (__bf16)fmaxf(acc, 0.0f);
  }
  __syncthreads();

  const int wave = t >> 5, lane = t & 31;
  const int hi = lane >> 4, ln = lane & 15;

  // ---- B fragments from LDS (same K order as A):
  //      lanes 0-15: N=lane, K 0..7 / 16..23 ; lanes 16-31: N=lane-16, K 8..15 / 24..31 ----
  union { uint4 q[2]; v16bf v; } b0u, b1u;
  const uint4* ehq = (const uint4*)(ehs + ln * 64);  // 8 uint4 per edge row
  b0u.q[0] = ehq[hi];
  b0u.q[1] = ehq[2 + hi];
  b1u.q[0] = ehq[4 + hi];
  b1u.q[1] = ehq[6 + hi];

  // ---- loop over 256 column tiles, 32 per wave ----
  for (int nt = wave; nt < 256; nt += 8) {
    const int c0 = nt * 16;
    const __bf16* ar = W2T + (size_t)(c0 + ln) * 64; // row of W2T for M=c0+ln
    // ISA 16-bit A 16x32 layout: lanes 0-15: K 0..7 / 16..23 ;
    //                            lanes 16-31: K 8..15 / 24..31
    union { uint4 q[2]; v16bf v; } a0u, a1u;
    a0u.q[0] = *(const uint4*)(ar + hi * 8);
    a0u.q[1] = *(const uint4*)(ar + 16 + hi * 8);
    a1u.q[0] = *(const uint4*)(ar + 32 + hi * 8);
    a1u.q[1] = *(const uint4*)(ar + 48 + hi * 8);

    v8f c;
#pragma unroll
    for (int r = 0; r < 8; ++r) c[r] = b2[c0 + r + hi * 8];  // bias on M

    c = __builtin_amdgcn_wmma_f32_16x16x32_bf16(false, a0u.v, false, b0u.v,
                                                (short)0, c, false, false);
    c = __builtin_amdgcn_wmma_f32_16x16x32_bf16(false, a1u.v, false, b1u.v,
                                                (short)0, c, false, false);

    // C: lane -> edge = e0 + ln ; VGPR r -> col = c0 + r + hi*8 (contiguous)
    union { __bf16 h[8]; uint4 q; } o;
#pragma unroll
    for (int r = 0; r < 8; ++r) o.h[r] = (__bf16)c[r];
    *(uint4*)(A + (size_t)(e0 + ln) * 4096 + c0 + hi * 8) = o.q;
  }
}

// --------------------------------------------------------------------------
// Message step: agg[tgt] += A[e] @ h[src].  One wave32 per edge; streams the
// 8 KB bf16 matrix at HBM roofline; lane l owns output rows l and l+32.
// --------------------------------------------------------------------------
__device__ __forceinline__ float bflo(unsigned u) { return __uint_as_float(u << 16); }
__device__ __forceinline__ float bfhi(unsigned u) { return __uint_as_float(u & 0xffff0000u); }

__device__ __forceinline__ float dot8(uint4 q, const float* __restrict__ hv) {
  float a = 0.0f;
  a = fmaf(bflo(q.x), hv[0], a);  a = fmaf(bfhi(q.x), hv[1], a);
  a = fmaf(bflo(q.y), hv[2], a);  a = fmaf(bfhi(q.y), hv[3], a);
  a = fmaf(bflo(q.z), hv[4], a);  a = fmaf(bfhi(q.z), hv[5], a);
  a = fmaf(bflo(q.w), hv[6], a);  a = fmaf(bfhi(q.w), hv[7], a);
  return a;
}

__global__ void __launch_bounds__(256) message_kernel(
    const __bf16* __restrict__ A, const float* __restrict__ h,
    const int* __restrict__ Esrc, const int* __restrict__ Etgt,
    float* __restrict__ agg) {
  __shared__ float hs[8][HDIM];
  const int wave = threadIdx.x >> 5, lane = threadIdx.x & 31;
  const int e = blockIdx.x * 8 + wave;               // 6250*8 == 50000 exactly
  const int src = Esrc[e], tgt = Etgt[e];

  hs[wave][lane]      = h[src * HDIM + lane];
  hs[wave][lane + 32] = h[src * HDIM + lane + 32];
  __syncthreads();

  const __bf16* Ae = A + (size_t)e * 4096;
  const uint4* row0 = (const uint4*)(Ae + lane * 64);
  const uint4* row1 = (const uint4*)(Ae + (lane + 32) * 64);
  float acc0 = 0.0f, acc1 = 0.0f;
#pragma unroll
  for (int kq = 0; kq < 8; ++kq) {
    uint4 q0 = row0[kq];
    uint4 q1 = row1[kq];
    float hv[8];
#pragma unroll
    for (int j = 0; j < 8; ++j) hv[j] = hs[wave][kq * 8 + j];
    acc0 += dot8(q0, hv);
    acc1 += dot8(q1, hv);
  }
  atomicAdd(&agg[(size_t)tgt * HDIM + lane],      acc0);
  atomicAdd(&agg[(size_t)tgt * HDIM + lane + 32], acc1);
}

// --------------------------------------------------------------------------
// GRU cell (fp32, torch gate layout r,z,n): 4 nodes per block, thread=(n,d).
// --------------------------------------------------------------------------
__global__ void __launch_bounds__(256) gru_kernel(
    const float* __restrict__ agg, const float* __restrict__ h,
    const float* __restrict__ Wih, const float* __restrict__ Whh,
    const float* __restrict__ bih, const float* __restrict__ bhh,
    float* __restrict__ hout) {
  __shared__ float xs[4][HDIM], hsh[4][HDIM];
  const int t = threadIdx.x;
  const int n0 = blockIdx.x * 4;
  xs[t >> 6][t & 63]  = agg[(size_t)n0 * HDIM + t];
  hsh[t >> 6][t & 63] = h[(size_t)n0 * HDIM + t];
  __syncthreads();

  const int ln = t >> 6, d = t & 63;
  const float* wri = Wih + (size_t)d * 64;
  const float* wzi = Wih + (size_t)(64 + d) * 64;
  const float* wni = Wih + (size_t)(128 + d) * 64;
  const float* wrh = Whh + (size_t)d * 64;
  const float* wzh = Whh + (size_t)(64 + d) * 64;
  const float* wnh = Whh + (size_t)(128 + d) * 64;

  float ir = bih[d], iz = bih[64 + d], in_ = bih[128 + d];
  float hr = bhh[d], hz = bhh[64 + d], hn = bhh[128 + d];
#pragma unroll 8
  for (int k = 0; k < HDIM; ++k) {
    float x = xs[ln][k], hh = hsh[ln][k];
    ir = fmaf(x, wri[k], ir);  iz = fmaf(x, wzi[k], iz);  in_ = fmaf(x, wni[k], in_);
    hr = fmaf(hh, wrh[k], hr); hz = fmaf(hh, wzh[k], hz); hn = fmaf(hh, wnh[k], hn);
  }
  float r = 1.0f / (1.0f + __expf(-(ir + hr)));
  float z = 1.0f / (1.0f + __expf(-(iz + hz)));
  float nn = tanhf(in_ + r * hn);
  hout[(size_t)n0 * HDIM + t] = (1.0f - z) * nn + z * hsh[ln][d];
}

// --------------------------------------------------------------------------
// out = h @ W_out + b_out ; pooled = segment_sum(out, batch)  [B,1]
// --------------------------------------------------------------------------
__global__ void __launch_bounds__(256) output_kernel(
    const float* __restrict__ h, const float* __restrict__ Wout,
    const float* __restrict__ bout, const int* __restrict__ batch,
    float* __restrict__ out) {
  int n = blockIdx.x * 256 + threadIdx.x;
  if (n >= N_NODES) return;
  float acc = bout[0];
  const float* hr = h + (size_t)n * HDIM;
#pragma unroll 8
  for (int k = 0; k < HDIM; ++k) acc = fmaf(hr[k], Wout[k], acc);
  atomicAdd(&out[batch[n]], acc);
}

// --------------------------------------------------------------------------
extern "C" void kernel_launch(void* const* d_in, const int* in_sizes, int n_in,
                              void* d_out, int out_size, void* d_ws, size_t ws_size,
                              hipStream_t stream) {
  const float* node_features = (const float*)d_in[0];
  const float* edge_features = (const float*)d_in[1];
  const int*   Esrc          = (const int*)d_in[2];
  const int*   Etgt          = (const int*)d_in[3];
  const int*   batch         = (const int*)d_in[4];
  const float* W_in  = (const float*)d_in[5];
  const float* b_in  = (const float*)d_in[6];
  const float* ee_W1 = (const float*)d_in[7];
  const float* ee_b1 = (const float*)d_in[8];
  const float* ee_W2 = (const float*)d_in[9];
  const float* ee_b2 = (const float*)d_in[10];
  const float* gWih  = (const float*)d_in[11];
  const float* gWhh  = (const float*)d_in[12];
  const float* gbih  = (const float*)d_in[13];
  const float* gbhh  = (const float*)d_in[14];
  const float* W_out = (const float*)d_in[15];
  const float* b_out = (const float*)d_in[16];

  // workspace layout (all 256B aligned)
  char* ws = (char*)d_ws;
  __bf16* A   = (__bf16*)(ws);                         // 50000*4096*2 = 409,600,000
  __bf16* W2T = (__bf16*)(ws + 409600000);             // 4096*64*2    =     524,288
  float*  hA  = (float*)(ws + 410124288);              // 12800*64*4   =   3,276,800
  float*  hB  = (float*)(ws + 413401088);              //              =   3,276,800
  float*  agg = (float*)(ws + 416677888);              //              =   3,276,800

  // one-time precompute
  w2t_kernel<<<1024, 256, 0, stream>>>(ee_W2, W2T);
  node_in_kernel<<<(N_NODES * HDIM) / 256, 256, 0, stream>>>(node_features, W_in, b_in, hA);
  edge_enc_kernel<<<N_EDGES / 16, 256, 0, stream>>>(edge_features, ee_W1, ee_b1,
                                                    W2T, ee_b2, A);

  float* hc = hA;
  float* hn = hB;
  for (int s = 0; s < NSTEPS; ++s) {
    hipMemsetAsync(agg, 0, (size_t)N_NODES * HDIM * sizeof(float), stream);
    message_kernel<<<N_EDGES / 8, 256, 0, stream>>>(A, hc, Esrc, Etgt, agg);
    gru_kernel<<<N_NODES / 4, 256, 0, stream>>>(agg, hc, gWih, gWhh, gbih, gbhh, hn);
    float* tmp = hc; hc = hn; hn = tmp;
  }

  hipMemsetAsync(d_out, 0, (size_t)BGRAPH * sizeof(float), stream);
  output_kernel<<<(N_NODES + 255) / 256, 256, 0, stream>>>(hc, W_out, b_out, batch,
                                                           (float*)d_out);
}
